// TransducerPredictor_72980084293742
// MI455X (gfx1250) — compile-verified, matrix operand order checked by reading, not run on hardware
//
#include <hip/hip_runtime.h>

// ---------------------------------------------------------------------------
// TransducerPredictor (2-layer LayerNormLSTM) for MI455X / gfx1250
// B=64, U=512, T=U+1=513, D=1024, 4D=4096, L=2, V=4096
// Recurrent scan runs as ONE persistent kernel per layer (64 WGs, grid-wide
// split barriers). Per step: WMMA recurrent GEMM overlapped with an async
// global->LDS copy of the step's xz row; fused LN/gate/LN cell phase.
// ---------------------------------------------------------------------------

#define B_   64
#define U_   512
#define T_   513
#define D_   1024
#define N4_  4096
#define L_   2
#define NWG_ 64
#define EPS_ 1e-3f

typedef _Float16 v8h  __attribute__((ext_vector_type(8)));
typedef _Float16 v16h __attribute__((ext_vector_type(16)));
typedef float    v8f  __attribute__((ext_vector_type(8)));

__device__ __forceinline__ float sigm_(float x) { return 1.0f / (1.0f + __expf(-x)); }
// tanh(x) = 1 - 2/(exp(2x)+1); single v_exp_f32
__device__ __forceinline__ float tanh_(float x) {
  return 1.0f - 2.0f / (__expf(2.0f * x) + 1.0f);
}

__device__ __forceinline__ v8f wmma_(v16h a, v16h b, v8f c) {
  return __builtin_amdgcn_wmma_f32_16x16x32_f16(false, a, false, b, (short)0, c,
                                                false, false);
}

// Load one 16-bit WMMA operand (16 halfs) for this lane from a K-major row:
// lane L(<16) holds K {k..k+7, k+16..k+23}, lane L+16 holds {k+8..15, k+24..31}
// caller passes p already offset by (lane>>4)*8.
__device__ __forceinline__ v16h ld_op_(const _Float16* p) {
  v16h r;
  ((v8h*)&r)[0] = *(const v8h*)(p);
  ((v8h*)&r)[1] = *(const v8h*)(p + 16);
  return r;
}

// ---- CDNA5 async global->LDS copy (ASYNCcnt path), inline asm form --------
// VDST = LDS byte address (low 32 bits of the generic shared pointer),
// VADDR = 64-bit global address.
__device__ __forceinline__ void async_g2l_b128_(unsigned lds_off,
                                                const void* gptr) {
  asm volatile("global_load_async_to_lds_b128 %0, %1, off"
               :: "v"(lds_off), "v"(gptr) : "memory");
}
__device__ __forceinline__ void wait_async0_() {
  asm volatile("s_wait_asynccnt 0x0" ::: "memory");
}
__device__ __forceinline__ unsigned lds_addr_(const void* p) {
  return (unsigned)(uintptr_t)p;  // LDS aperture: low 32 bits = LDS byte offset
}

// ---------------------------------------------------------------------------
// Grid-wide split barrier (generation counting). Release fence before arrive,
// acquire fence after release, executed by EVERY thread so each wave
// invalidates its own L0 (waves of a WG can be on either CU of the WGP).
// ---------------------------------------------------------------------------
__device__ __forceinline__ void grid_barrier_(unsigned* cnt, unsigned* gen,
                                              unsigned nwg) {
  __builtin_amdgcn_fence(__ATOMIC_RELEASE, "agent");
  __syncthreads();
  if (threadIdx.x == 0) {
    unsigned g0 = __atomic_load_n(gen, __ATOMIC_RELAXED);
    unsigned a  = __atomic_fetch_add(cnt, 1u, __ATOMIC_RELAXED);
    if (a + 1u == nwg) {
      __atomic_store_n(cnt, 0u, __ATOMIC_RELAXED);
      __atomic_fetch_add(gen, 1u, __ATOMIC_RELEASE);
    } else {
      while (__atomic_load_n(gen, __ATOMIC_RELAXED) == g0)
        __builtin_amdgcn_s_sleep(2);
    }
  }
  __syncthreads();
  __builtin_amdgcn_fence(__ATOMIC_ACQUIRE, "agent");
}

// ---------------------------------------------------------------------------
// Weight transpose + f32->f16 convert:  W [K, N] f32  ->  Wt [N, K] f16
// ---------------------------------------------------------------------------
__global__ __launch_bounds__(256) void transpose_w_kernel(
    const float* __restrict__ W, _Float16* __restrict__ Wt, int K, int N) {
  size_t idx = (size_t)blockIdx.x * blockDim.x + threadIdx.x;
  if (idx >= (size_t)K * N) return;
  int k = (int)(idx / N);
  int n = (int)(idx % N);
  Wt[(size_t)n * K + k] = (_Float16)W[idx];
}

// ---------------------------------------------------------------------------
// Embedding gather + SOS frame:  x[b,0,:]=0 ; x[b,t,:]=E[targets[b,t-1],:]
// ---------------------------------------------------------------------------
__global__ __launch_bounds__(256) void embed_kernel(
    const int* __restrict__ targets, const float* __restrict__ E,
    _Float16* __restrict__ x) {
  size_t idx = (size_t)blockIdx.x * blockDim.x + threadIdx.x;
  if (idx >= (size_t)B_ * T_ * D_) return;
  int d = (int)(idx % D_);
  int t = (int)((idx / D_) % T_);
  int b = (int)(idx / ((size_t)D_ * T_));
  float v = 0.0f;
  if (t > 0) {
    int tok = targets[b * U_ + (t - 1)];
    v = E[(size_t)tok * D_ + d];
  }
  x[idx] = (_Float16)v;
}

// ---------------------------------------------------------------------------
// Bulk WMMA GEMM (input projection): C[M,N] f16 = A[M,K] f16 * Bt[N,K] f16
// Each wave32: 32x32 output tile = 2x2 v_wmma_f32_16x16x32_f16, K-loop of 32.
// ---------------------------------------------------------------------------
__global__ __launch_bounds__(256) void wmma_gemm_f16_kernel(
    const _Float16* __restrict__ A, const _Float16* __restrict__ Bt,
    _Float16* __restrict__ C, int M, int N, int K) {
  int wave = (int)((blockIdx.x * blockDim.x + threadIdx.x) >> 5);
  int lane = threadIdx.x & 31;
  int wavesN = N >> 5;
  int wm = wave / wavesN;
  int wn = wave % wavesN;
  if (wm * 32 >= M) return;  // wave-uniform

  const int r0 = wm * 32, c0 = wn * 32;
  const int lrow = lane & 15;
  const int koff = (lane >> 4) * 8;

  v8f acc[2][2] = {};
  for (int k0 = 0; k0 < K; k0 += 32) {
    v16h a[2], b[2];
#pragma unroll
    for (int i = 0; i < 2; ++i) {
      a[i] = ld_op_(A + (size_t)(r0 + i * 16 + lrow) * K + k0 + koff);
      b[i] = ld_op_(Bt + (size_t)(c0 + i * 16 + lrow) * K + k0 + koff);
    }
#pragma unroll
    for (int i = 0; i < 2; ++i)
#pragma unroll
      for (int j = 0; j < 2; ++j)
        acc[i][j] = wmma_(a[i], b[j], acc[i][j]);
  }
  const int crow = (lane >> 4) * 8;
  const int ccol = lane & 15;
#pragma unroll
  for (int i = 0; i < 2; ++i)
#pragma unroll
    for (int j = 0; j < 2; ++j)
#pragma unroll
      for (int v = 0; v < 8; ++v)
        C[(size_t)(r0 + i * 16 + crow + v) * N + (c0 + j * 16 + ccol)] =
            (_Float16)acc[i][j][v];
}

// ---------------------------------------------------------------------------
// Row LayerNorm (in-place on f16 buffer), one workgroup per row of length F.
// ---------------------------------------------------------------------------
__global__ __launch_bounds__(256) void ln_rows_kernel(
    _Float16* __restrict__ xz, const float* __restrict__ g,
    const float* __restrict__ bparm, int F) {
  int tid = threadIdx.x;
  _Float16* xr = xz + (size_t)blockIdx.x * F;
  __shared__ float red[256];
  __shared__ float st[2];

  float s = 0.0f, s2 = 0.0f;
  for (int j = tid; j < F; j += 256) {
    float v = (float)xr[j];
    s += v; s2 += v * v;
  }
  red[tid] = s; __syncthreads();
  for (int off = 128; off > 0; off >>= 1) {
    if (tid < off) red[tid] += red[tid + off];
    __syncthreads();
  }
  if (tid == 0) st[0] = red[0] / (float)F;
  __syncthreads();
  red[tid] = s2; __syncthreads();
  for (int off = 128; off > 0; off >>= 1) {
    if (tid < off) red[tid] += red[tid + off];
    __syncthreads();
  }
  if (tid == 0) st[1] = rsqrtf(red[0] / (float)F - st[0] * st[0] + EPS_);
  __syncthreads();
  float m = st[0], rs = st[1];
  for (int j = tid; j < F; j += 256)
    xr[j] = (_Float16)(((float)xr[j] - m) * rs * g[j] + bparm[j]);
}

// ---------------------------------------------------------------------------
// Persistent recurrent scan for one layer. Grid = 64 WGs x 256 threads.
// Per step:
//   Phase A: zpre[64,4096] = h_cur[64,1024] @ WrT; WG g owns cols [g*64,..+64),
//            8 waves in 2(M)x4(N) grid of 32x16 WMMA tiles. Concurrently the
//            step's xz row is async-copied global->LDS, and the next step's
//            row is prefetched into L2.
//   barrier ; s_wait_asynccnt 0 ; syncthreads
//   Phase B: WG b = cell update of batch row b; c state lives in registers,
//            xz row read from LDS.
//   barrier
// ---------------------------------------------------------------------------
__global__ __launch_bounds__(256) void lstm_scan_kernel(
    const _Float16* __restrict__ xz,    // [B,T,4D] LN'd input projection
    const _Float16* __restrict__ wrT,   // [4D, D]
    const float* __restrict__ bias, const float* __restrict__ gr,
    const float* __restrict__ br, const float* __restrict__ gc,
    const float* __restrict__ bc,
    float* __restrict__ zpre,           // [B,4D] scratch
    _Float16* __restrict__ h_cur,       // [B,D], pre-zeroed
    _Float16* __restrict__ h_seq,       // [B,T,D] or null
    float* __restrict__ out,            // [B,D,T] or null
    unsigned* __restrict__ bar_cnt, unsigned* __restrict__ bar_gen) {
  const int wg   = blockIdx.x;          // N-chunk in phase A, batch row in B
  const int tid  = threadIdx.x;
  const int wave = tid >> 5;
  const int lane = tid & 31;

  // Phase A tile assignment: 2(M) x 4(N) waves, each 32x16 output tile
  const int wm   = wave & 1;
  const int wn   = wave >> 1;
  const int r0   = wm * 32;
  const int c0   = wg * 64 + wn * 16;
  const int lrow = lane & 15;
  const int koff = (lane >> 4) * 8;
  const int crow = (lane >> 4) * 8;
  const int ccol = lane & 15;

  // Persistent cell state for batch row `wg`: d = tid + 256*q
  float c_reg[4] = {0.0f, 0.0f, 0.0f, 0.0f};

  __shared__ _Float16 xs[N4_];          // staged xz row (8 KB)
  __shared__ float red[256];
  __shared__ float st[2];

  for (int t = 0; t < T_; ++t) {
    // ---- kick off async global->LDS copy of this step's xz row ----
    {
      const _Float16* xrow = xz + ((size_t)wg * T_ + t) * N4_;
      async_g2l_b128_(lds_addr_(&xs[tid * 8]), xrow + tid * 8);
      async_g2l_b128_(lds_addr_(&xs[2048 + tid * 8]), xrow + 2048 + tid * 8);
      if (t + 1 < T_) {  // L2 prefetch of next step's row (global_prefetch_b8)
        const char* nrow = (const char*)(xz + ((size_t)wg * T_ + t + 1) * N4_);
        __builtin_prefetch(nrow + tid * 32, 0, 1);
      }
    }

    // ------------------ Phase A: recurrent WMMA GEMM ------------------
    {
      v8f acc0 = {}, acc1 = {};
      const _Float16* bbase = wrT + (size_t)(c0 + lrow) * D_ + koff;
      const _Float16* abase = h_cur + (size_t)(r0 + lrow) * D_ + koff;
      for (int k0 = 0; k0 < D_; k0 += 32) {
        v16h a0 = ld_op_(abase + k0);
        v16h a1 = ld_op_(abase + (size_t)16 * D_ + k0);
        v16h bv = ld_op_(bbase + k0);
        acc0 = wmma_(a0, bv, acc0);
        acc1 = wmma_(a1, bv, acc1);
      }
#pragma unroll
      for (int v = 0; v < 8; ++v) {
        zpre[(size_t)(r0 + crow + v) * N4_ + (c0 + ccol)]      = acc0[v];
        zpre[(size_t)(r0 + 16 + crow + v) * N4_ + (c0 + ccol)] = acc1[v];
      }
    }
    grid_barrier_(bar_cnt, bar_gen, NWG_);
    wait_async0_();      // per-wave ASYNCcnt drain ...
    __syncthreads();     // ... then WG-wide: xs fully populated

    // ------------------ Phase B: fused LSTM cell ------------------
    {
      const float* zr = zpre + (size_t)wg * N4_;

      // LN stats over zpre row (4096)
      float s = 0.0f, s2 = 0.0f;
#pragma unroll
      for (int q = 0; q < 16; ++q) {
        float v = zr[tid + 256 * q];
        s += v; s2 += v * v;
      }
      red[tid] = s; __syncthreads();
      for (int off = 128; off > 0; off >>= 1) {
        if (tid < off) red[tid] += red[tid + off];
        __syncthreads();
      }
      if (tid == 0) st[0] = red[0] / (float)N4_;
      __syncthreads();
      red[tid] = s2; __syncthreads();
      for (int off = 128; off > 0; off >>= 1) {
        if (tid < off) red[tid] += red[tid + off];
        __syncthreads();
      }
      if (tid == 0) st[1] = rsqrtf(red[0] / (float)N4_ - st[0] * st[0] + EPS_);
      __syncthreads();
      float m = st[0], rs = st[1];

      // gates + cell update; thread owns d = tid + 256*q; xz row from LDS
      float cn[4], zo_s[4];
      float cs = 0.0f, cs2 = 0.0f;
#pragma unroll
      for (int q = 0; q < 4; ++q) {
        int d = tid + 256 * q;
        int ji = d, jf = D_ + d, jg = 2 * D_ + d, jo = 3 * D_ + d;
        float zi = (float)xs[ji] + (zr[ji] - m) * rs * gr[ji] + br[ji] + bias[ji];
        float zf = (float)xs[jf] + (zr[jf] - m) * rs * gr[jf] + br[jf] + bias[jf];
        float zg = (float)xs[jg] + (zr[jg] - m) * rs * gr[jg] + br[jg] + bias[jg];
        float zo = (float)xs[jo] + (zr[jo] - m) * rs * gr[jo] + br[jo] + bias[jo];
        float cc = sigm_(zf) * c_reg[q] + sigm_(zi) * tanh_(zg);
        cn[q] = cc; zo_s[q] = zo;
        cs += cc; cs2 += cc * cc;
      }
      __syncthreads();

      // LN stats over new c (1024)
      red[tid] = cs; __syncthreads();
      for (int off = 128; off > 0; off >>= 1) {
        if (tid < off) red[tid] += red[tid + off];
        __syncthreads();
      }
      if (tid == 0) st[0] = red[0] / (float)D_;
      __syncthreads();
      red[tid] = cs2; __syncthreads();
      for (int off = 128; off > 0; off >>= 1) {
        if (tid < off) red[tid] += red[tid + off];
        __syncthreads();
      }
      if (tid == 0) st[1] = rsqrtf(red[0] / (float)D_ - st[0] * st[0] + EPS_);
      __syncthreads();
      float m2 = st[0], rs2 = st[1];

#pragma unroll
      for (int q = 0; q < 4; ++q) {
        int d = tid + 256 * q;
        float cl = (cn[q] - m2) * rs2 * gc[d] + bc[d];
        c_reg[q] = cl;
        float h = sigm_(zo_s[q]) * tanh_(cl);
        h_cur[(size_t)wg * D_ + d] = (_Float16)h;
        if (h_seq) h_seq[((size_t)wg * T_ + t) * D_ + d] = (_Float16)h;
        if (out) out[((size_t)wg * D_ + d) * T_ + t] = h;
      }
    }
    grid_barrier_(bar_cnt, bar_gen, NWG_);
  }
}

// ---------------------------------------------------------------------------
// Host orchestration
// ---------------------------------------------------------------------------
extern "C" void kernel_launch(void* const* d_in, const int* in_sizes, int n_in,
                              void* d_out, int out_size, void* d_ws, size_t ws_size,
                              hipStream_t stream) {
  const int*   targets = (const int*)d_in[0];
  const float* E       = (const float*)d_in[1];
  const float* Wk      = (const float*)d_in[2];   // [L, D, 4D]
  const float* Wr      = (const float*)d_in[3];   // [L, D, 4D]
  const float* bias    = (const float*)d_in[4];   // [L, 4D]
  const float* gk      = (const float*)d_in[5];
  const float* bk      = (const float*)d_in[6];
  const float* gr      = (const float*)d_in[7];
  const float* br      = (const float*)d_in[8];
  const float* gc      = (const float*)d_in[9];   // [L, D]
  const float* bc      = (const float*)d_in[10];
  float* out = (float*)d_out;                      // [B, D, T]

  // ---- workspace carve (256B aligned) ----
  char* w = (char*)d_ws;
  auto carve = [&](size_t bytes) -> void* {
    void* p = (void*)w;
    w += (bytes + 255) & ~(size_t)255;
    return p;
  };
  const size_t wsz = (size_t)N4_ * D_;
  _Float16* wkT   = (_Float16*)carve(L_ * wsz * sizeof(_Float16));
  _Float16* wrT   = (_Float16*)carve(L_ * wsz * sizeof(_Float16));
  _Float16* xbuf  = (_Float16*)carve((size_t)B_ * T_ * D_ * sizeof(_Float16));
  _Float16* hbuf  = (_Float16*)carve((size_t)B_ * T_ * D_ * sizeof(_Float16));
  _Float16* xz    = (_Float16*)carve((size_t)B_ * T_ * N4_ * sizeof(_Float16));
  float*    zpre  = (float*)carve((size_t)B_ * N4_ * sizeof(float));
  _Float16* h_cur = (_Float16*)carve((size_t)B_ * D_ * sizeof(_Float16));
  unsigned* bars  = (unsigned*)carve(2 * sizeof(unsigned));  // {cnt, gen}

  (void)hipMemsetAsync(bars, 0, 2 * sizeof(unsigned), stream);

  // ---- one-time weight transpose/convert (f32 [K,N] -> f16 [N,K]) ----
  {
    int blocks = (int)((wsz + 255) / 256);
    for (int l = 0; l < L_; ++l) {
      transpose_w_kernel<<<blocks, 256, 0, stream>>>(Wk + l * wsz, wkT + l * wsz, D_, N4_);
      transpose_w_kernel<<<blocks, 256, 0, stream>>>(Wr + l * wsz, wrT + l * wsz, D_, N4_);
    }
  }

  // ---- embedding + SOS ----
  {
    size_t tot = (size_t)B_ * T_ * D_;
    embed_kernel<<<(int)((tot + 255) / 256), 256, 0, stream>>>(targets, E, xbuf);
  }

  const int BT = B_ * T_;
  const int gemm_in_blocks = (BT / 32) * (N4_ / 32) / 8;  // 8 waves / block

  for (int l = 0; l < L_; ++l) {
    const _Float16* xin = (l == 0) ? xbuf : hbuf;
    _Float16* hseq = (l == 0) ? hbuf : nullptr;
    float*    oout = (l == 0) ? nullptr : out;

    // xz = x @ Wk (f16), then in-place kernel LayerNorm
    wmma_gemm_f16_kernel<<<gemm_in_blocks, 256, 0, stream>>>(
        xin, wkT + l * wsz, xz, BT, N4_, D_);
    ln_rows_kernel<<<BT, 256, 0, stream>>>(xz, gk + l * N4_, bk + l * N4_, N4_);

    // reset recurrent operand, then run the whole scan in one persistent kernel
    (void)hipMemsetAsync(h_cur, 0, (size_t)B_ * D_ * sizeof(_Float16), stream);
    lstm_scan_kernel<<<NWG_, 256, 0, stream>>>(
        xz, wrT + l * wsz, bias + l * N4_, gr + l * N4_, br + l * N4_,
        gc + l * D_, bc + l * D_, zpre, h_cur, hseq, oout, bars, bars + 1);
  }
}
